// GPTneo_42408507081092
// MI455X (gfx1250) — matrix operand order repaired and checked by reading, not compiled, not run on hardware
//
#include <hip/hip_runtime.h>
#include <hip/hip_bf16.h>
#include <math.h>

// ---------------------------------------------------------------------------
// GPT-2-ish forward for MI455X (gfx1250): all GEMMs via v_wmma_f32_16x16x32_bf16
// L=12, D=768, H=12, HD=64, S=2048, F=3072, V=50257, B=1
// GEMM: 128x128/block (8 waves, 4M x 2N), 32x64/wave = 2x4 WMMA tiles,
// K-step 32, LDS double-buffered (ping-pong), one barrier per k-step.
// ---------------------------------------------------------------------------

#define GPT_L  12
#define GPT_D  768
#define GPT_H  12
#define GPT_HD 64
#define GPT_S  2048
#define GPT_F  3072
#define GPT_V  50257

#define BM 128
#define BN 128
#define BK 32

typedef __bf16 v16bf __attribute__((ext_vector_type(16)));
typedef float  v8f   __attribute__((ext_vector_type(8)));

__device__ __forceinline__ unsigned short f2bf_bits(float f) {
    union { float f; unsigned int u; } c; c.f = f;
    unsigned int r = c.u + 0x7FFFu + ((c.u >> 16) & 1u);   // RNE
    return (unsigned short)(r >> 16);
}

__device__ __forceinline__ float gelu_exact(float x) {
    return 0.5f * x * (1.0f + erff(x * 0.70710678118654752f));
}

// Stage one (A 128x32, B 32x128) tile pair into LDS as bf16.
//   As row m holds K in natural order [0-31] -> lane fragment is one
//   contiguous 32B read.  Bs is transposed (Bs[n][k]) -> lane fragment is one
//   contiguous 32B read.  bvec selects float4 B staging (ldb % 4 == 0).
__device__ __forceinline__ void stage_AB(
    const float* __restrict__ A, int lda,
    const float* __restrict__ Bm, int ldb,
    int blockM, int blockN, int N, int kk,
    unsigned short* __restrict__ As, unsigned short* __restrict__ Bs,
    int tid, bool bvec) {
#pragma unroll
    for (int i4 = tid; i4 < (BM * BK) / 4; i4 += 256) {
        int m  = i4 >> 3;               // 0..127
        int kg = (i4 & 7) << 2;         // 0,4,...,28
        const float4 av =
            *(const float4*)(A + (size_t)(blockM + m) * lda + kk + kg);
        int g    = kg >> 3;
        int slot = g;                                    // 0,1,2,3
        int base = m * BK + slot * 8 + (kg & 7);
        As[base + 0] = f2bf_bits(av.x);
        As[base + 1] = f2bf_bits(av.y);
        As[base + 2] = f2bf_bits(av.z);
        As[base + 3] = f2bf_bits(av.w);
    }
    if (bvec) {  // vectorized B staging (all GEMMs except LM head)
#pragma unroll
        for (int i4 = tid; i4 < (BN * BK) / 4; i4 += 256) {
            int k  = i4 >> 5;               // 0..31
            int ng = (i4 & 31) << 2;        // 0,4,...,124
            int gn = blockN + ng;
            float4 bv;
            if (gn + 3 < N) {
                bv = *(const float4*)(Bm + (size_t)(kk + k) * ldb + gn);
            } else {
                const float* row = Bm + (size_t)(kk + k) * ldb;
                bv.x = (gn + 0 < N) ? row[gn + 0] : 0.0f;
                bv.y = (gn + 1 < N) ? row[gn + 1] : 0.0f;
                bv.z = (gn + 2 < N) ? row[gn + 2] : 0.0f;
                bv.w = (gn + 3 < N) ? row[gn + 3] : 0.0f;
            }
            Bs[(ng + 0) * BK + k] = f2bf_bits(bv.x);
            Bs[(ng + 1) * BK + k] = f2bf_bits(bv.y);
            Bs[(ng + 2) * BK + k] = f2bf_bits(bv.z);
            Bs[(ng + 3) * BK + k] = f2bf_bits(bv.w);
        }
    } else {     // scalar fallback (ldb = V = 50257)
#pragma unroll
        for (int i = tid; i < BN * BK; i += 256) {
            int k  = i >> 7;
            int n  = i & 127;
            int gn = blockN + n;
            float v = (gn < N) ? Bm[(size_t)(kk + k) * ldb + gn] : 0.0f;
            Bs[n * BK + k] = f2bf_bits(v);
        }
    }
}

// ---------------------------------------------------------------------------
// C[M,N] = act(A[M,K] @ B[K,N] + bias) + R      (act: 0=none, 1=exact GELU)
// Requires M % 128 == 0 and K % 32 == 0 (true at every call site).
// ---------------------------------------------------------------------------
__global__ __launch_bounds__(256)
void gemm_bf16_wmma(const float* __restrict__ A, int lda,
                    const float* __restrict__ Bm, int ldb,
                    const float* __restrict__ bias,
                    const float* __restrict__ R, int ldr,
                    float* __restrict__ C, int ldc,
                    int M, int N, int K, int act) {
    const int tid   = threadIdx.x;
    const int lane  = tid & 31;
    const int wave  = tid >> 5;
    const int waveM = wave >> 1;
    const int waveN = wave & 1;
    const int l15   = lane & 15;
    const int half  = lane >> 4;

    const int blockM = blockIdx.y * BM;
    const int blockN = blockIdx.x * BN;
    const int mBase  = waveM * 32;
    const int nBase  = waveN * 64;
    const bool bvec  = ((ldb & 3) == 0);

    __shared__ __align__(64) unsigned short As[2][BM * BK];
    __shared__ __align__(64) unsigned short Bs[2][BN * BK];

    v8f zero{};
    v8f acc[2][4];
#pragma unroll
    for (int i = 0; i < 2; ++i)
#pragma unroll
        for (int j = 0; j < 4; ++j) acc[i][j] = zero;

    const int nsteps = K / BK;

    // prologue: stage first tile
    stage_AB(A, lda, Bm, ldb, blockM, blockN, N, 0, As[0], Bs[0], tid, bvec);
    __syncthreads();

    for (int it = 0; it < nsteps; ++it) {
        const int kk  = it * BK;
        const int cur = it & 1;

        // overlap: stage next tile into the other buffer while computing
        if (it + 1 < nsteps)
            stage_AB(A, lda, Bm, ldb, blockM, blockN, N, kk + BK,
                     As[cur ^ 1], Bs[cur ^ 1], tid, bvec);
        if (it + 2 < nsteps) {
            __builtin_prefetch(A + (size_t)(blockM + mBase) * lda + kk + 2 * BK, 0, 1);
            __builtin_prefetch(Bm + (size_t)(kk + 2 * BK) * ldb + blockN, 0, 1);
        }

        // fragments from current buffer: one 32B LDS read each
        v16bf afr[2], bfr[4];
#pragma unroll
        for (int i = 0; i < 2; ++i) {
            const unsigned short* p =
                &As[cur][(mBase + i * 16 + l15) * BK + half * 16];
            afr[i] = *(const v16bf*)p;
        }
#pragma unroll
        for (int j = 0; j < 4; ++j) {
            const unsigned short* p =
                &Bs[cur][(nBase + j * 16 + l15) * BK + half * 16];
            bfr[j] = *(const v16bf*)p;
        }

        // 8 WMMAs per wave per k-step
#pragma unroll
        for (int i = 0; i < 2; ++i)
#pragma unroll
            for (int j = 0; j < 4; ++j)
                acc[i][j] = __builtin_amdgcn_wmma_f32_16x16x32_bf16(
                    false, afr[i], false, bfr[j], (short)0, acc[i][j],
                    false, false);

        __syncthreads();   // next-buffer stores + current-buffer reads done
    }

    // epilogue: C/D layout -> lane = N col, VGPR r = M row (+8 upper half)
#pragma unroll
    for (int j = 0; j < 4; ++j) {
        int n = blockN + nBase + j * 16 + l15;
        if (n >= N) continue;
        float bv = bias ? bias[n] : 0.0f;
#pragma unroll
        for (int i = 0; i < 2; ++i) {
#pragma unroll
            for (int r = 0; r < 8; ++r) {
                int m = blockM + mBase + i * 16 + half * 8 + r;
                float v = acc[i][j][r] + bv;
                if (act == 1) v = gelu_exact(v);
                if (R) v += R[(size_t)m * ldr + n];
                C[(size_t)m * ldc + n] = v;
            }
        }
    }
}

// ---------------------------------------------------------------------------
// Row LayerNorm: one block per row of [rows, D]
// ---------------------------------------------------------------------------
__global__ __launch_bounds__(256)
void layernorm_rows(const float* __restrict__ in,
                    const float* __restrict__ g,
                    const float* __restrict__ b,
                    float* __restrict__ out, int D) {
    const int row = blockIdx.x;
    const float* x = in + (size_t)row * D;
    float* y = out + (size_t)row * D;
    __shared__ float red[256];
    const int tid = threadIdx.x;

    float s = 0.0f;
    for (int i = tid; i < D; i += 256) s += x[i];
    red[tid] = s; __syncthreads();
    for (int off = 128; off > 0; off >>= 1) {
        if (tid < off) red[tid] += red[tid + off];
        __syncthreads();
    }
    const float mu = red[0] / (float)D;
    __syncthreads();

    float vs = 0.0f;
    for (int i = tid; i < D; i += 256) { float d = x[i] - mu; vs += d * d; }
    red[tid] = vs; __syncthreads();
    for (int off = 128; off > 0; off >>= 1) {
        if (tid < off) red[tid] += red[tid + off];
        __syncthreads();
    }
    const float rstd = rsqrtf(red[0] / (float)D + 1e-5f);
    __syncthreads();

    for (int i = tid; i < D; i += 256)
        y[i] = (x[i] - mu) * rstd * g[i] + b[i];
}

// ---------------------------------------------------------------------------
// Causal softmax over [S, S] scores in-place (additive -1e38 mask, no scale)
// ---------------------------------------------------------------------------
__global__ __launch_bounds__(256)
void softmax_causal(float* __restrict__ sc, int S) {
    const int row = blockIdx.x;
    float* p = sc + (size_t)row * S;
    __shared__ float red[256];
    const int tid = threadIdx.x;

    float mx = -3.4e38f;
    for (int c = tid; c < S; c += 256) {
        float v = p[c] + ((c > row) ? -1e38f : 0.0f);
        mx = fmaxf(mx, v);
    }
    red[tid] = mx; __syncthreads();
    for (int off = 128; off > 0; off >>= 1) {
        if (tid < off) red[tid] = fmaxf(red[tid], red[tid + off]);
        __syncthreads();
    }
    mx = red[0];
    __syncthreads();

    float sum = 0.0f;
    for (int c = tid; c < S; c += 256) {
        float v = p[c] + ((c > row) ? -1e38f : 0.0f);
        float e = __expf(v - mx);
        p[c] = e;
        sum += e;
    }
    red[tid] = sum; __syncthreads();
    for (int off = 128; off > 0; off >>= 1) {
        if (tid < off) red[tid] += red[tid + off];
        __syncthreads();
    }
    const float inv = 1.0f / red[0];
    __syncthreads();

    for (int c = tid; c < S; c += 256) p[c] *= inv;
}

// ---------------------------------------------------------------------------
// Transpose [S, D] -> [D, S] (coalesced read)
// ---------------------------------------------------------------------------
__global__ __launch_bounds__(256)
void transpose_sd(const float* __restrict__ in, float* __restrict__ out,
                  int S, int D) {
    int idx = blockIdx.x * 256 + threadIdx.x;
    if (idx < S * D) {
        int s = idx / D, d = idx - s * D;
        out[(size_t)d * S + s] = in[idx];
    }
}

// ---------------------------------------------------------------------------
// Host side
// ---------------------------------------------------------------------------
static inline void launch_gemm(const float* A, int lda, const float* B, int ldb,
                               const float* bias, const float* R, int ldr,
                               float* C, int ldc, int M, int N, int K, int act,
                               hipStream_t st) {
    dim3 grid((N + BN - 1) / BN, M / BM);  // M % 128 == 0 at every call site
    gemm_bf16_wmma<<<grid, dim3(256), 0, st>>>(A, lda, B, ldb, bias, R, ldr,
                                               C, ldc, M, N, K, act);
}

extern "C" void kernel_launch(void* const* d_in, const int* in_sizes, int n_in,
                              void* d_out, int out_size, void* d_ws, size_t ws_size,
                              hipStream_t stream) {
    (void)in_sizes; (void)n_in; (void)out_size; (void)ws_size;

    const int S = GPT_S, D = GPT_D, F = GPT_F, V = GPT_V, H = GPT_H, HD = GPT_HD;

    const float* x     = (const float*)d_in[0];   // [1,S,D]
    const float* ln1_g = (const float*)d_in[1];   // [L,D]
    const float* ln1_b = (const float*)d_in[2];
    const float* Wq    = (const float*)d_in[3];   // [L,D,D]
    const float* Wk    = (const float*)d_in[4];
    const float* Wv    = (const float*)d_in[5];
    const float* Wo    = (const float*)d_in[6];
    const float* bo    = (const float*)d_in[7];   // [L,D]
    const float* ln2_g = (const float*)d_in[8];
    const float* ln2_b = (const float*)d_in[9];
    const float* Wfc   = (const float*)d_in[10];  // [L,D,F]
    const float* bfc   = (const float*)d_in[11];  // [L,F]
    const float* Wp    = (const float*)d_in[12];  // [L,F,D]
    const float* bp    = (const float*)d_in[13];  // [L,D]
    const float* lnf_g = (const float*)d_in[14];  // [D]
    const float* lnf_b = (const float*)d_in[15];
    const float* Wlm   = (const float*)d_in[16];  // [D,V]
    float* out = (float*)d_out;                   // [S,V]

    // workspace layout (floats), ~80 MB total
    float* ws = (float*)d_ws;
    const size_t SD = (size_t)S * D;
    float* h      = ws;                  // [S,D] hidden state
    float* hn     = h      + SD;         // [S,D] layernorm output
    float* q      = hn     + SD;         // [S,D]
    float* kT     = q      + SD;         // [D,S] transposed keys
    float* v      = kT     + SD;         // [S,D]
    float* attnb  = v      + SD;         // [S,D] attention output
    float* ff1    = attnb  + SD;         // [S,F]
    float* scores = ff1 + (size_t)S * F; // [S,S] (also used as k temp)

    hipMemcpyAsync(h, x, SD * sizeof(float), hipMemcpyDeviceToDevice, stream);

    for (int l = 0; l < GPT_L; ++l) {
        const float* wq  = Wq  + (size_t)l * D * D;
        const float* wk  = Wk  + (size_t)l * D * D;
        const float* wv  = Wv  + (size_t)l * D * D;
        const float* wo  = Wo  + (size_t)l * D * D;
        const float* wfc = Wfc + (size_t)l * D * F;
        const float* wp  = Wp  + (size_t)l * F * D;

        layernorm_rows<<<S, 256, 0, stream>>>(h, ln1_g + l * D, ln1_b + l * D, hn, D);

        // q/k/v projections (no biases in reference)
        launch_gemm(hn, D, wq, D, nullptr, nullptr, 0, q, D, S, D, D, 0, stream);
        launch_gemm(hn, D, wk, D, nullptr, nullptr, 0, scores /*ktmp*/, D, S, D, D, 0, stream);
        launch_gemm(hn, D, wv, D, nullptr, nullptr, 0, v, D, S, D, D, 0, stream);

        // kT[D,S]; head hh occupies rows hh*HD .. hh*HD+63
        transpose_sd<<<((S * D) + 255) / 256, 256, 0, stream>>>(scores, kT, S, D);

        for (int hh = 0; hh < H; ++hh) {
            // scores = q_h [S,HD] @ kT_h [HD,S]
            launch_gemm(q + hh * HD, D, kT + (size_t)hh * HD * S, S,
                        nullptr, nullptr, 0, scores, S, S, S, HD, 0, stream);
            softmax_causal<<<S, 256, 0, stream>>>(scores, S);
            // attn_h = probs [S,S] @ v_h [S,HD]
            launch_gemm(scores, S, v + hh * HD, D,
                        nullptr, nullptr, 0, attnb + hh * HD, D, S, HD, S, 0, stream);
        }

        // h = h + attn @ Wo + bo
        launch_gemm(attnb, D, wo, D, bo + l * D, h, D, h, D, S, D, D, 0, stream);

        layernorm_rows<<<S, 256, 0, stream>>>(h, ln2_g + l * D, ln2_b + l * D, hn, D);

        // ff1 = gelu(hn @ Wfc + bfc)
        launch_gemm(hn, D, wfc, F, bfc + l * F, nullptr, 0, ff1, F, S, F, D, 1, stream);

        // h = h + ff1 @ Wp + bp
        launch_gemm(ff1, F, wp, D, bp + l * D, h, D, h, D, S, D, F, 0, stream);
    }

    layernorm_rows<<<S, 256, 0, stream>>>(h, lnf_g, lnf_b, hn, D);
    launch_gemm(hn, D, Wlm, V, nullptr, nullptr, 0, out, V, S, V, D, 0, stream);
}
